// SelfAttention_41807211659827
// MI455X (gfx1250) — matrix operand order repaired and checked by reading, not compiled, not run on hardware
//
#include <hip/hip_runtime.h>

// ---------------------------------------------------------------------------
// Self-attention (b=4, t=2048, d=1024, h=16, hd=64) for gfx1250 (MI455X).
// - All GEMMs on v_wmma_f32_16x16x32_bf16 (f32 accumulate), softmax scale
//   d^-0.5 = 1/32 folded into Q.
// - Projection GEMMs: 128x128 block tiles, A/B staged into LDS by the Tensor
//   Data Mover (tensor_load_to_lds, TENSORcnt), double-buffered; 8 waves each
//   compute a 32x64 subtile from LDS fragments.
// - Attention: flash-style streaming softmax, one wave per 16 query rows.
// Workspace ~92 MB of bf16 staging.
// ---------------------------------------------------------------------------

#define DIM   1024
#define TLEN  2048
#define BATCH 4
#define HEADS 16
#define HD    64
#define BT    (BATCH * TLEN)        // 8192
#define BH    (BATCH * HEADS)       // 64

typedef __bf16 bf16_t;
typedef bf16_t v16bf __attribute__((ext_vector_type(16)));
typedef float  v8f   __attribute__((ext_vector_type(8)));
typedef unsigned int u32x4 __attribute__((ext_vector_type(4)));
typedef int          i32x4 __attribute__((ext_vector_type(4)));
typedef int          i32x8 __attribute__((ext_vector_type(8)));

union FragU { v16bf v; uint4 q[2]; };

#if __has_builtin(__builtin_amdgcn_tensor_load_to_lds)
#define HAVE_TDM 1
#if __has_include(<hip/amd_detail/amd_gfx1250_TDM.h>)
#define TDM_6ARG 1
#endif
#endif

// f32 -> bf16 round-to-nearest-even
static __device__ __forceinline__ unsigned short f2bf(float f) {
    unsigned u = __float_as_uint(f);
    u += 0x7FFFu + ((u >> 16) & 1u);
    return (unsigned short)(u >> 16);
}

// A-operand fragment (16x32 bf16) from row-major [rows x ld] ushort storage.
// lane<16: row=lane, K 0..7 & 16..23 ; lane>=16: row=lane-16, K 8..15 & 24..31
static __device__ __forceinline__ v16bf load_fragA(const unsigned short* base, int ld, int lane) {
    int half = lane >> 4, lrow = lane & 15;
    const unsigned short* p = base + lrow * ld + half * 8;
    FragU f;
    f.q[0] = *(const uint4*)(p);
    f.q[1] = *(const uint4*)(p + 16);
    return f.v;
}

// B-operand fragment (32x16 bf16) from transposed-B storage (each B column is
// a contiguous row of length ld). lane<16: col=lane, K 0..7 & 16..23 ; lane>=16: K 8..15 & 24..31
static __device__ __forceinline__ v16bf load_fragB(const unsigned short* base, int ld, int lane) {
    int half = lane >> 4, lrow = lane & 15;
    const unsigned short* p = base + lrow * ld + half * 8;
    FragU f;
    f.q[0] = *(const uint4*)(p);
    f.q[1] = *(const uint4*)(p + 16);
    return f.v;
}

static __device__ __forceinline__ v8f wmma_bf16(v16bf a, v16bf b, v8f c) {
    return __builtin_amdgcn_wmma_f32_16x16x32_bf16(false, a, false, b, (short)0, c, false, false);
}

#ifdef HAVE_TDM
// Issue a TDM 2-D tile load: global row-major bf16 [.. x ld_elems] -> LDS,
// tile = tile_rows x tile_cols elements, packed row-major (row pitch tile_cols).
// D# layout per CDNA5 ISA (group0 128b, group1 256b).
static __device__ __forceinline__ void tdm_load_2d(const unsigned short* gptr,
                                                   unsigned lds_byte_addr,
                                                   unsigned tile_cols,
                                                   unsigned tile_rows,
                                                   unsigned ld_elems,
                                                   unsigned tensor_rows) {
    unsigned long long ga = (unsigned long long)(const void*)gptr;
    u32x4 g0;
    g0[0] = 1u;                                            // count=1 (valid user D#)
    g0[1] = lds_byte_addr;                                 // lds_addr
    g0[2] = (unsigned)(ga & 0xFFFFFFFFu);                  // global_addr[31:0]
    g0[3] = (unsigned)((ga >> 32) & 0x01FFFFFFu) | (2u << 30); // [56:32] | type=2
    i32x8 g1;
    unsigned long long s0 = ld_elems;                      // tensor_dim0_stride
    g1[0] = (int)(1u << 16);                               // wg_mask=0, data_size=1 (2B)
    g1[1] = (int)((ld_elems & 0xFFFFu) << 16);             // tensor_dim0[15:0]
    g1[2] = (int)((ld_elems >> 16) | ((tensor_rows & 0xFFFFu) << 16)); // dim0 hi | dim1 lo
    g1[3] = (int)((tensor_rows >> 16) | (tile_cols << 16));// dim1 hi | tile_dim0
    g1[4] = (int)(tile_rows & 0xFFFFu);                    // tile_dim1 | tile_dim2=0
    g1[5] = (int)(s0 & 0xFFFFFFFFull);                     // dim0_stride lo32
    g1[6] = (int)((s0 >> 32) & 0xFFFFull);                 // dim0_stride hi16 | dim1_stride lo=0
    g1[7] = 0;                                             // dim1_stride hi (2-D: unused)
    i32x4 z4 = {0, 0, 0, 0};
#ifdef TDM_6ARG
    i32x8 z8 = {0, 0, 0, 0, 0, 0, 0, 0};
    __builtin_amdgcn_tensor_load_to_lds(g0, g1, z4, z4, z8, 0);
#else
    __builtin_amdgcn_tensor_load_to_lds(g0, g1, z4, z4, 0);
#endif
}
#endif

// Fill one 128x32 bf16 LDS tile (row-major, pitch 32) from global [.. x ld].
static __device__ __forceinline__ void stage_tile(unsigned short* lds,
                                                  const unsigned short* g,
                                                  int ld, int tid, bool issuer) {
#ifdef HAVE_TDM
    (void)tid;
    if (issuer)
        tdm_load_2d(g, (unsigned)(unsigned long long)(void*)lds, 32u, 128u,
                    (unsigned)ld, 0xFFFFu);
#else
    (void)issuer;
    // cooperative fallback: 512 x uint4 (8 bf16 each), 256 threads, 2 rounds
#pragma unroll
    for (int rep = 0; rep < 2; ++rep) {
        int idx = tid + rep * 256;           // 0..511
        int row = idx >> 2;                  // /4 uint4 per row
        int c8  = (idx & 3) * 8;
        *(uint4*)(lds + row * 32 + c8) = *(const uint4*)(g + (size_t)row * ld + c8);
    }
#endif
}

static __device__ __forceinline__ void stage_wait(bool issuer) {
#ifdef HAVE_TDM
    if (issuer) __builtin_amdgcn_s_wait_tensorcnt(0);
#else
    (void)issuer;
#endif
    __syncthreads();
}

// ---------------------------------------------------------------------------
__global__ void k_cvt(const float* __restrict__ in, unsigned short* __restrict__ out, int n) {
    int i = blockIdx.x * blockDim.x + threadIdx.x;
    if (i < n) out[i] = f2bf(in[i]);
}

// in: [K x N] f32 row-major -> out: [N x K] bf16 row-major (transpose+convert)
__global__ void k_cvt_t(const float* __restrict__ in, unsigned short* __restrict__ out, int K, int N) {
    int i = blockIdx.x * blockDim.x + threadIdx.x;
    if (i < K * N) {
        int k = i / N, n = i % N;
        out[(size_t)n * K + k] = f2bf(in[i]);
    }
}

// ---------------------------------------------------------------------------
// QKV GEMM: block computes a 128(M) x 128(N) tile of [BT x 3*DIM]; A/B tiles
// double-buffered in LDS via TDM; 8 waves x (32x64) subtiles; epilogue writes
// q (pre-scaled 1/32), k row-major and v transposed, head-major bf16.
__global__ void __launch_bounds__(256)
k_gemm_qkv(const unsigned short* __restrict__ Xb,
           const unsigned short* __restrict__ WqkvT,   // [3*DIM][DIM]
           unsigned short* __restrict__ Qs,            // [BH][T][HD]
           unsigned short* __restrict__ Ks,            // [BH][T][HD]
           unsigned short* __restrict__ Vt) {          // [BH][HD][T]
    __shared__ unsigned short ldsA[2][128 * 32];
    __shared__ unsigned short ldsB[2][128 * 32];
    const int NBLK = (3 * DIM) / 128;                  // 24
    int m0 = (blockIdx.x / NBLK) * 128;
    int n0 = (blockIdx.x % NBLK) * 128;
    int tid  = threadIdx.x;
    int w    = tid >> 5;
    int lane = tid & 31;
    int half = lane >> 4, lrow = lane & 15;
    int mw = (w >> 1) * 32;                            // wave row offset in tile
    int nw = (w & 1) * 64;                             // wave col offset in tile
    bool issuer = (w == 0);

    stage_tile(ldsA[0], Xb    + (size_t)m0 * DIM, DIM, tid, issuer);
    stage_tile(ldsB[0], WqkvT + (size_t)n0 * DIM, DIM, tid, issuer);
    stage_wait(issuer);

    v8f acc[2][4] = {};
    for (int k0 = 0; k0 < DIM; k0 += 32) {
        int cur = (k0 >> 5) & 1;
        bool more = (k0 + 32) < DIM;
        if (more) {
            stage_tile(ldsA[cur ^ 1], Xb    + (size_t)m0 * DIM + k0 + 32, DIM, tid, issuer);
            stage_tile(ldsB[cur ^ 1], WqkvT + (size_t)n0 * DIM + k0 + 32, DIM, tid, issuer);
        }
        v16bf a0 = load_fragA(ldsA[cur] + (mw + 0)  * 32, 32, lane);
        v16bf a1 = load_fragA(ldsA[cur] + (mw + 16) * 32, 32, lane);
#pragma unroll
        for (int n = 0; n < 4; ++n) {
            v16bf b = load_fragB(ldsB[cur] + (nw + n * 16) * 32, 32, lane);
            acc[0][n] = wmma_bf16(a0, b, acc[0][n]);
            acc[1][n] = wmma_bf16(a1, b, acc[1][n]);
        }
        if (more) stage_wait(issuer);
    }

    // epilogue: wave's 64-col span (n0+nw ..+63) is one of q/k/v and one head
    int nbase = n0 + nw;
    int sel   = nbase / DIM;                           // 0=q 1=k 2=v
    int head  = (nbase % DIM) / HD;
    int bb    = m0 / TLEN;                             // 128-row block within batch
    int t0    = m0 % TLEN;
    int bh    = bb * HEADS + head;
#pragma unroll
    for (int mi = 0; mi < 2; ++mi) {
#pragma unroll
        for (int n = 0; n < 4; ++n) {
#pragma unroll
            for (int r = 0; r < 8; ++r) {
                float val = acc[mi][n][r];
                int t   = t0 + mw + mi * 16 + r + 8 * half;
                int col = n * 16 + lrow;               // hd index 0..63
                if (sel == 0)
                    Qs[(size_t)bh * TLEN * HD + (size_t)t * HD + col] = f2bf(val * 0.03125f);
                else if (sel == 1)
                    Ks[(size_t)bh * TLEN * HD + (size_t)t * HD + col] = f2bf(val);
                else
                    Vt[(size_t)bh * HD * TLEN + (size_t)col * TLEN + t] = f2bf(val);
            }
        }
    }
}

// ---------------------------------------------------------------------------
// Flash attention: one wave handles 16 query rows of one (b,h); streams over
// keys/values in 32-column steps with online softmax.  LDS stages P (bf16)
// to convert C-layout scores into an A-layout operand for the PV WMMA.
__global__ void __launch_bounds__(128)
k_attn(const unsigned short* __restrict__ Qs,
       const unsigned short* __restrict__ Ks,
       const unsigned short* __restrict__ Vt,
       unsigned short* __restrict__ Ab) {              // [BT][DIM] bf16
    __shared__ unsigned short pbuf[4][16 * 32];
    int wslot = threadIdx.x >> 5;
    int wave  = blockIdx.x * (blockDim.x >> 5) + wslot;
    const int ROWBLK = TLEN / 16;                      // 128
    int bh   = wave / ROWBLK;
    int m0   = (wave % ROWBLK) * 16;
    int lane = threadIdx.x & 31;
    int half = lane >> 4, lrow = lane & 15;

    const unsigned short* Qbase = Qs + (size_t)bh * TLEN * HD + (size_t)m0 * HD;
    const unsigned short* Kbase = Ks + (size_t)bh * TLEN * HD;
    const unsigned short* Vbase = Vt + (size_t)bh * HD * TLEN;

    v16bf qa0 = load_fragA(Qbase + 0,  HD, lane);      // d = 0..31
    v16bf qa1 = load_fragA(Qbase + 32, HD, lane);      // d = 32..63

    v8f o[4] = {};
    float mrow[8], lsum[8];
#pragma unroll
    for (int r = 0; r < 8; ++r) { mrow[r] = -3.0e38f; lsum[r] = 0.0f; }

    unsigned short* pb = pbuf[wslot];

    for (int j0 = 0; j0 < TLEN; j0 += 32) {
        v8f s0 = {}, s1 = {};
        s0 = wmma_bf16(qa0, load_fragB(Kbase + (size_t)(j0     ) * HD + 0,  HD, lane), s0);
        s0 = wmma_bf16(qa1, load_fragB(Kbase + (size_t)(j0     ) * HD + 32, HD, lane), s0);
        s1 = wmma_bf16(qa0, load_fragB(Kbase + (size_t)(j0 + 16) * HD + 0,  HD, lane), s1);
        s1 = wmma_bf16(qa1, load_fragB(Kbase + (size_t)(j0 + 16) * HD + 32, HD, lane), s1);

        float mnew[8], corr[8];
#pragma unroll
        for (int r = 0; r < 8; ++r) {
            float tmax = fmaxf(s0[r], s1[r]);
            tmax = fmaxf(tmax, __shfl_xor(tmax, 1, 32));
            tmax = fmaxf(tmax, __shfl_xor(tmax, 2, 32));
            tmax = fmaxf(tmax, __shfl_xor(tmax, 4, 32));
            tmax = fmaxf(tmax, __shfl_xor(tmax, 8, 32));
            mnew[r] = fmaxf(mrow[r], tmax);
            corr[r] = __expf(mrow[r] - mnew[r]);
            mrow[r] = mnew[r];
        }
#pragma unroll
        for (int r = 0; r < 8; ++r) {
            float p0 = __expf(s0[r] - mnew[r]);
            float p1 = __expf(s1[r] - mnew[r]);
            s0[r] = p0; s1[r] = p1;
            float rs = p0 + p1;
            rs += __shfl_xor(rs, 1, 32);
            rs += __shfl_xor(rs, 2, 32);
            rs += __shfl_xor(rs, 4, 32);
            rs += __shfl_xor(rs, 8, 32);
            lsum[r] = lsum[r] * corr[r] + rs;
        }
#pragma unroll
        for (int n = 0; n < 4; ++n)
#pragma unroll
            for (int r = 0; r < 8; ++r)
                o[n][r] *= corr[r];

        // P (C-layout) -> LDS 16x32 bf16; DS ops of one wave are in-order.
#pragma unroll
        for (int r = 0; r < 8; ++r) {
            int row = r + 8 * half;
            pb[row * 32 + lrow]      = f2bf(s0[r]);
            pb[row * 32 + 16 + lrow] = f2bf(s1[r]);
        }
        asm volatile("" ::: "memory");
        FragU pf;
        {
            const unsigned short* pp = pb + lrow * 32 + half * 8;   // A-layout gather
            pf.q[0] = *(const uint4*)(pp);
            pf.q[1] = *(const uint4*)(pp + 16);
        }
        asm volatile("" ::: "memory");

#pragma unroll
        for (int n = 0; n < 4; ++n) {
            v16bf vb = load_fragB(Vbase + (size_t)(n * 16) * TLEN + j0, TLEN, lane);
            o[n] = wmma_bf16(pf.v, vb, o[n]);
        }
    }

    int bb = bh / HEADS, h = bh % HEADS;
#pragma unroll
    for (int r = 0; r < 8; ++r) {
        float inv = 1.0f / lsum[r];
        int row = m0 + r + 8 * half;
#pragma unroll
        for (int n = 0; n < 4; ++n) {
            int col = h * HD + n * 16 + lrow;
            Ab[(size_t)(bb * TLEN + row) * DIM + col] = f2bf(o[n][r] * inv);
        }
    }
}

// ---------------------------------------------------------------------------
// Output projection: Y = Ab @ W_out + b_out (f32 out), same TDM-staged tiling.
__global__ void __launch_bounds__(256)
k_gemm_out(const unsigned short* __restrict__ Ab,
           const unsigned short* __restrict__ WoutT,   // [DIM][DIM] (N-major)
           const float* __restrict__ bias,
           float* __restrict__ Y) {
    __shared__ unsigned short ldsA[2][128 * 32];
    __shared__ unsigned short ldsB[2][128 * 32];
    const int NBLK = DIM / 128;                        // 8
    int m0 = (blockIdx.x / NBLK) * 128;
    int n0 = (blockIdx.x % NBLK) * 128;
    int tid  = threadIdx.x;
    int w    = tid >> 5;
    int lane = tid & 31;
    int half = lane >> 4, lrow = lane & 15;
    int mw = (w >> 1) * 32;
    int nw = (w & 1) * 64;
    bool issuer = (w == 0);

    stage_tile(ldsA[0], Ab    + (size_t)m0 * DIM, DIM, tid, issuer);
    stage_tile(ldsB[0], WoutT + (size_t)n0 * DIM, DIM, tid, issuer);
    stage_wait(issuer);

    v8f acc[2][4] = {};
    for (int k0 = 0; k0 < DIM; k0 += 32) {
        int cur = (k0 >> 5) & 1;
        bool more = (k0 + 32) < DIM;
        if (more) {
            stage_tile(ldsA[cur ^ 1], Ab    + (size_t)m0 * DIM + k0 + 32, DIM, tid, issuer);
            stage_tile(ldsB[cur ^ 1], WoutT + (size_t)n0 * DIM + k0 + 32, DIM, tid, issuer);
        }
        v16bf a0 = load_fragA(ldsA[cur] + (mw + 0)  * 32, 32, lane);
        v16bf a1 = load_fragA(ldsA[cur] + (mw + 16) * 32, 32, lane);
#pragma unroll
        for (int n = 0; n < 4; ++n) {
            v16bf b = load_fragB(ldsB[cur] + (nw + n * 16) * 32, 32, lane);
            acc[0][n] = wmma_bf16(a0, b, acc[0][n]);
            acc[1][n] = wmma_bf16(a1, b, acc[1][n]);
        }
        if (more) stage_wait(issuer);
    }

#pragma unroll
    for (int mi = 0; mi < 2; ++mi) {
#pragma unroll
        for (int n = 0; n < 4; ++n) {
#pragma unroll
            for (int r = 0; r < 8; ++r) {
                int row = m0 + mw + mi * 16 + r + 8 * half;
                int col = n0 + nw + n * 16 + lrow;
                Y[(size_t)row * DIM + col] = acc[mi][n][r] + bias[col];
            }
        }
    }
}

// ---------------------------------------------------------------------------
extern "C" void kernel_launch(void* const* d_in, const int* in_sizes, int n_in,
                              void* d_out, int out_size, void* d_ws, size_t ws_size,
                              hipStream_t stream) {
    const float* X    = (const float*)d_in[0];   // [4,2048,1024]
    const float* Wqkv = (const float*)d_in[1];   // [1024,3072]
    const float* Wout = (const float*)d_in[2];   // [1024,1024]
    const float* bout = (const float*)d_in[3];   // [1024]
    float* Y = (float*)d_out;

    char*  ws  = (char*)d_ws;
    size_t off = 0;
    auto alloc = [&](size_t bytes) -> void* {
        void* p = ws + off;
        off += (bytes + 255) & ~(size_t)255;
        return p;
    };
    unsigned short* Xb    = (unsigned short*)alloc((size_t)BT * DIM * 2);       // 16 MB
    unsigned short* WqkvT = (unsigned short*)alloc((size_t)3 * DIM * DIM * 2);  //  6 MB
    unsigned short* WoutT = (unsigned short*)alloc((size_t)DIM * DIM * 2);      //  2 MB
    unsigned short* Qs    = (unsigned short*)alloc((size_t)BH * TLEN * HD * 2); // 16 MB
    unsigned short* Ks    = (unsigned short*)alloc((size_t)BH * TLEN * HD * 2); // 16 MB
    unsigned short* Vt    = (unsigned short*)alloc((size_t)BH * HD * TLEN * 2); // 16 MB
    unsigned short* Ab    = (unsigned short*)alloc((size_t)BT * DIM * 2);       // 16 MB
    (void)ws_size; (void)in_sizes; (void)n_in; (void)out_size;

    k_cvt  <<<(BT * DIM + 255) / 256,      256, 0, stream>>>(X, Xb, BT * DIM);
    k_cvt_t<<<(3 * DIM * DIM + 255) / 256, 256, 0, stream>>>(Wqkv, WqkvT, DIM, 3 * DIM);
    k_cvt_t<<<(DIM * DIM + 255) / 256,     256, 0, stream>>>(Wout, WoutT, DIM, DIM);

    // QKV GEMM: (8192/128) x (3072/128) = 64 x 24 = 1536 blocks of 256 threads
    k_gemm_qkv<<<(BT / 128) * (3 * DIM / 128), 256, 0, stream>>>(Xb, WqkvT, Qs, Ks, Vt);

    // Attention: 64 (b,h) x 128 row-blocks = 8192 waves, 4 waves/block
    k_attn<<<BH * (TLEN / 16) / 4, 128, 0, stream>>>(Qs, Ks, Vt, Ab);

    // Output projection: (8192/128) x (1024/128) = 64 x 8 = 512 blocks
    k_gemm_out<<<(BT / 128) * (DIM / 128), 256, 0, stream>>>(Ab, WoutT, bout, Y);
}
